// GraphSAGEClassifier_31653908972164
// MI455X (gfx1250) — matrix operand order repaired and checked by reading, not compiled, not run on hardware
//
#include <hip/hip_runtime.h>

// ---------------------------------------------------------------------------
// GraphSAGE classifier for MI455X (gfx1250, wave32, WMMA bf16 -> f32)
// ---------------------------------------------------------------------------
#define NNODES 100000
#define NEDGES 3200000
#define DIM    256
#define NCLS   47
#define EPB    256          // edges per gather block (3.2M / 256 = 12500)
#define ROWS   16           // output rows per GEMM block (100000/16 = 6250)

typedef __attribute__((ext_vector_type(16))) __bf16 bf16x16;
typedef __attribute__((ext_vector_type(8)))  __bf16 bf16x8;
typedef __attribute__((ext_vector_type(8)))  float  f32x8;

__device__ __forceinline__ unsigned short f2bf(float f) {
    unsigned int u = __float_as_uint(f);
    u += 0x7FFFu + ((u >> 16) & 1u);   // round-to-nearest-even
    return (unsigned short)(u >> 16);
}

// A-fragment (16x32 bf16, M x K) from an LDS strip of 16 rows x 256 cols.
// Contiguous k order used here: lane<16 -> M=lane, K = kk..kk+15
//             lane>=16 -> M=lane-16, K = kk+16..kk+31
__device__ __forceinline__ bf16x16 load_a_frag(const unsigned short* sA, int lane, int kk) {
    int m    = lane & 15;
    int base = kk + ((lane >> 4) << 4);
    bf16x8 lo = *(const bf16x8*)(sA + m * DIM + base);
    bf16x8 hi = *(const bf16x8*)(sA + m * DIM + base + 8);
    bf16x16 a;
#pragma unroll
    for (int i = 0; i < 8; i++) { a[i] = lo[i]; a[i + 8] = hi[i]; }
    return a;
}

// B-fragment (32x16 bf16, K x N) for out = h @ W^T: B column j == row j of W.
// Same contiguous k order as A: lane<16 -> N=lane, K=kk..kk+15 ; lane>=16 -> N=lane-16, K=kk+16..kk+31
// => one contiguous 32-byte load per lane from row-major bf16 W.
__device__ __forceinline__ bf16x16 load_b_frag(const unsigned short* __restrict__ W,
                                               int lane, int jt, int kk) {
    int j  = jt + (lane & 15);
    int k0 = kk + ((lane >> 4) << 4);
    return *(const bf16x16*)(W + (size_t)j * DIM + k0);
}

// f32 -> bf16 conversion with zero padding (for W_post padded to 48 rows)
__global__ __launch_bounds__(256)
void k_cvt(const float* __restrict__ src, unsigned short* __restrict__ dst,
           int n_src, int n_dst) {
    int i = blockIdx.x * 256 + threadIdx.x;
    if (i < n_dst) dst[i] = (i < n_src) ? f2bf(src[i]) : (unsigned short)0;
}

// ---------------------------------------------------------------------------
// Sorted-dst segment-sum gather:  m[d,:] += h[s,:]
// 256 threads <-> 256 features; 256 edges per block; run-length accumulate,
// flush with global_atomic_add_f32 at segment boundaries.
// ---------------------------------------------------------------------------
__global__ __launch_bounds__(256)
void k_gather(const float* __restrict__ H, const int* __restrict__ src,
              const int* __restrict__ dst, float* __restrict__ M) {
    __shared__ int sS[EPB];
    __shared__ int sD[EPB];
    int tid = threadIdx.x;
    size_t e0 = (size_t)blockIdx.x * EPB;
    sS[tid] = src[e0 + tid];
    sD[tid] = dst[e0 + tid];
    __syncthreads();

    float acc = 0.0f;
    int prev = sD[0];
#pragma unroll 4
    for (int e = 0; e < EPB; e++) {
        int d = sD[e];
        if (d != prev) {
            atomicAdd(&M[(size_t)prev * DIM + tid], acc);
            acc = 0.0f;
            prev = d;
        }
        acc += H[(size_t)sS[e] * DIM + tid];
    }
    atomicAdd(&M[(size_t)prev * DIM + tid], acc);
}

// ---------------------------------------------------------------------------
// Pre-layer: out = l2norm(relu(X @ W^T + b))  (relu BEFORE norm, per reference)
// 128 threads = 4 waves; each wave owns 4 of the 16 column tiles.
// ---------------------------------------------------------------------------
__global__ __launch_bounds__(128)
void k_gemm_pre(const float* __restrict__ X, const unsigned short* __restrict__ Wb,
                const float* __restrict__ bias, float* __restrict__ Out) {
    __shared__ unsigned short sA[ROWS * DIM];
    __shared__ float sOut[ROWS * DIM];
    __shared__ float sRed[ROWS * 8];
    __shared__ float sInv[ROWS];
    const int tid  = threadIdx.x;
    const size_t row0 = (size_t)blockIdx.x * ROWS;

#pragma unroll
    for (int i = 0; i < 32; i++) {
        int idx = i * 128 + tid;
        sA[idx] = f2bf(X[row0 * DIM + idx]);
    }
    __syncthreads();

    const int wave = tid >> 5, lane = tid & 31;
    f32x8 acc[4];
#pragma unroll
    for (int c = 0; c < 4; c++) { f32x8 z = {}; acc[c] = z; }

#pragma unroll
    for (int ks = 0; ks < 8; ks++) {
        int kk = ks * 32;
        bf16x16 a = load_a_frag(sA, lane, kk);
#pragma unroll
        for (int c = 0; c < 4; c++) {
            bf16x16 b = load_b_frag(Wb, lane, (wave * 4 + c) * 16, kk);
            acc[c] = __builtin_amdgcn_wmma_f32_16x16x32_bf16(
                false, a, false, b, (short)0, acc[c], false, false);
        }
    }

    // epilogue: bias + relu into LDS (C/D layout: lane<16 -> M=v, lane>=16 -> M=v+8)
#pragma unroll
    for (int c = 0; c < 4; c++) {
        int col   = (wave * 4 + c) * 16 + (lane & 15);
        int rbase = (lane >> 4) << 3;
        float bj  = bias[col];
#pragma unroll
        for (int v = 0; v < 8; v++)
            sOut[(rbase + v) * DIM + col] = fmaxf(acc[c][v] + bj, 0.0f);
    }
    __syncthreads();

    { // row l2 norm (8 threads per row)
        int r = tid >> 3, p = tid & 7;
        float s = 0.0f;
        for (int c2 = 0; c2 < 32; c2++) { float v = sOut[r * DIM + p * 32 + c2]; s += v * v; }
        sRed[r * 8 + p] = s;
    }
    __syncthreads();
    if ((tid & 7) == 0) {
        int r = tid >> 3; float s = 0.0f;
        for (int i = 0; i < 8; i++) s += sRed[r * 8 + i];
        sInv[r] = 1.0f / fmaxf(sqrtf(s), 1e-12f);
    }
    __syncthreads();
#pragma unroll
    for (int i = 0; i < 32; i++) {
        int idx = i * 128 + tid;
        Out[row0 * DIM + idx] = sOut[idx] * sInv[idx >> 8];
    }
}

// ---------------------------------------------------------------------------
// SAGE conv: z = M @ Wl^T + bl + H @ Wr^T ; z = relu(l2norm(z)); H <- z (in place)
// ---------------------------------------------------------------------------
__global__ __launch_bounds__(128)
void k_gemm_conv(float* __restrict__ H, const float* __restrict__ Msum,
                 const unsigned short* __restrict__ Wl,
                 const unsigned short* __restrict__ Wr,
                 const float* __restrict__ bias) {
    __shared__ unsigned short sM[ROWS * DIM];
    __shared__ unsigned short sH[ROWS * DIM];
    __shared__ float sOut[ROWS * DIM];
    __shared__ float sRed[ROWS * 8];
    __shared__ float sInv[ROWS];
    const int tid  = threadIdx.x;
    const size_t row0 = (size_t)blockIdx.x * ROWS;

#pragma unroll
    for (int i = 0; i < 32; i++) {
        int idx = i * 128 + tid;
        sM[idx] = f2bf(Msum[row0 * DIM + idx]);
        sH[idx] = f2bf(H[row0 * DIM + idx]);
    }
    __syncthreads();

    const int wave = tid >> 5, lane = tid & 31;
    f32x8 acc[4];
#pragma unroll
    for (int c = 0; c < 4; c++) { f32x8 z = {}; acc[c] = z; }

#pragma unroll
    for (int ks = 0; ks < 8; ks++) {
        int kk = ks * 32;
        bf16x16 am = load_a_frag(sM, lane, kk);
        bf16x16 ah = load_a_frag(sH, lane, kk);
#pragma unroll
        for (int c = 0; c < 4; c++) {
            int jt = (wave * 4 + c) * 16;
            bf16x16 bl = load_b_frag(Wl, lane, jt, kk);
            acc[c] = __builtin_amdgcn_wmma_f32_16x16x32_bf16(
                false, am, false, bl, (short)0, acc[c], false, false);
            bf16x16 br = load_b_frag(Wr, lane, jt, kk);
            acc[c] = __builtin_amdgcn_wmma_f32_16x16x32_bf16(
                false, ah, false, br, (short)0, acc[c], false, false);
        }
    }

#pragma unroll
    for (int c = 0; c < 4; c++) {
        int col   = (wave * 4 + c) * 16 + (lane & 15);
        int rbase = (lane >> 4) << 3;
        float bj  = bias[col];
#pragma unroll
        for (int v = 0; v < 8; v++)
            sOut[(rbase + v) * DIM + col] = acc[c][v] + bj;   // no relu before norm
    }
    __syncthreads();

    {
        int r = tid >> 3, p = tid & 7;
        float s = 0.0f;
        for (int c2 = 0; c2 < 32; c2++) { float v = sOut[r * DIM + p * 32 + c2]; s += v * v; }
        sRed[r * 8 + p] = s;
    }
    __syncthreads();
    if ((tid & 7) == 0) {
        int r = tid >> 3; float s = 0.0f;
        for (int i = 0; i < 8; i++) s += sRed[r * 8 + i];
        sInv[r] = 1.0f / fmaxf(sqrtf(s), 1e-12f);
    }
    __syncthreads();
#pragma unroll
    for (int i = 0; i < 32; i++) {
        int idx = i * 128 + tid;
        H[row0 * DIM + idx] = fmaxf(sOut[idx] * sInv[idx >> 8], 0.0f);  // relu after norm
    }
}

// ---------------------------------------------------------------------------
// Final: t = relu(H @ Wjk^T + bjk); y = t @ Wpost^T + bpost; log_softmax(y)
// Wpost is pre-padded to 48x256 bf16 (row 47 = zeros).
// ---------------------------------------------------------------------------
__global__ __launch_bounds__(128)
void k_final(const float* __restrict__ H, const unsigned short* __restrict__ Wjk,
             const float* __restrict__ bjk, const unsigned short* __restrict__ Wpost,
             const float* __restrict__ bpost, float* __restrict__ Out) {
    __shared__ unsigned short sA[ROWS * DIM];
    __shared__ float sMid[ROWS * DIM];
    __shared__ unsigned short sT[ROWS * DIM];
    __shared__ float sP[ROWS * 48];
    const int tid  = threadIdx.x;
    const size_t row0 = (size_t)blockIdx.x * ROWS;
    const int wave = tid >> 5, lane = tid & 31;

#pragma unroll
    for (int i = 0; i < 32; i++) {
        int idx = i * 128 + tid;
        sA[idx] = f2bf(H[row0 * DIM + idx]);
    }
    __syncthreads();

    // stage 1: jk GEMM (bias + relu)
    f32x8 acc[4];
#pragma unroll
    for (int c = 0; c < 4; c++) { f32x8 z = {}; acc[c] = z; }
#pragma unroll
    for (int ks = 0; ks < 8; ks++) {
        int kk = ks * 32;
        bf16x16 a = load_a_frag(sA, lane, kk);
#pragma unroll
        for (int c = 0; c < 4; c++) {
            bf16x16 b = load_b_frag(Wjk, lane, (wave * 4 + c) * 16, kk);
            acc[c] = __builtin_amdgcn_wmma_f32_16x16x32_bf16(
                false, a, false, b, (short)0, acc[c], false, false);
        }
    }
#pragma unroll
    for (int c = 0; c < 4; c++) {
        int col   = (wave * 4 + c) * 16 + (lane & 15);
        int rbase = (lane >> 4) << 3;
        float bj  = bjk[col];
#pragma unroll
        for (int v = 0; v < 8; v++)
            sMid[(rbase + v) * DIM + col] = fmaxf(acc[c][v] + bj, 0.0f);
    }
    __syncthreads();
#pragma unroll
    for (int i = 0; i < 32; i++) {
        int idx = i * 128 + tid;
        sT[idx] = f2bf(sMid[idx]);
    }
    __syncthreads();

    // stage 2: post GEMM (48 cols -> waves 0..2 take one 16-col tile each)
    if (wave < 3) {
        f32x8 pa; { f32x8 z = {}; pa = z; }
        int jt = wave * 16;
#pragma unroll
        for (int ks = 0; ks < 8; ks++) {
            int kk = ks * 32;
            bf16x16 a = load_a_frag(sT, lane, kk);
            bf16x16 b = load_b_frag(Wpost, lane, jt, kk);
            pa = __builtin_amdgcn_wmma_f32_16x16x32_bf16(
                false, a, false, b, (short)0, pa, false, false);
        }
        int col   = jt + (lane & 15);
        int rbase = (lane >> 4) << 3;
        float bj  = (col < NCLS) ? bpost[col] : 0.0f;
#pragma unroll
        for (int v = 0; v < 8; v++)
            sP[(rbase + v) * 48 + col] = pa[v] + bj;
    }
    __syncthreads();

    // log_softmax per row over 47 classes
    if (tid < ROWS) {
        int r = tid;
        float mx = -1e30f;
        for (int j = 0; j < NCLS; j++) mx = fmaxf(mx, sP[r * 48 + j]);
        float s = 0.0f;
        for (int j = 0; j < NCLS; j++) s += expf(sP[r * 48 + j] - mx);
        float ls = logf(s);
        for (int j = 0; j < NCLS; j++)
            Out[(row0 + r) * NCLS + j] = sP[r * 48 + j] - mx - ls;
    }
}

// ---------------------------------------------------------------------------
extern "C" void kernel_launch(void* const* d_in, const int* in_sizes, int n_in,
                              void* d_out, int out_size, void* d_ws, size_t ws_size,
                              hipStream_t stream) {
    const float* x      = (const float*)d_in[0];
    const int*   src    = (const int*)d_in[1];
    const int*   dst    = (const int*)d_in[2];
    const float* W_pre  = (const float*)d_in[3];
    const float* b_pre  = (const float*)d_in[4];
    const float* Wl1    = (const float*)d_in[5];
    const float* bl1    = (const float*)d_in[6];
    const float* Wr1    = (const float*)d_in[7];
    const float* Wl2    = (const float*)d_in[8];
    const float* bl2    = (const float*)d_in[9];
    const float* Wr2    = (const float*)d_in[10];
    const float* W_jk   = (const float*)d_in[11];
    const float* b_jk   = (const float*)d_in[12];
    const float* W_post = (const float*)d_in[13];
    const float* b_post = (const float*)d_in[14];

    const size_t actBytes = (size_t)NNODES * DIM * sizeof(float);
    char* ws = (char*)d_ws;
    float* h = (float*)ws;                       // activations (updated in place)
    float* m = (float*)(ws + actBytes);          // segment-sum buffer
    unsigned short* wb = (unsigned short*)(ws + 2 * actBytes);
    const int WSZ = DIM * DIM;                   // 65536 elems per square weight
    unsigned short* wb_pre  = wb;
    unsigned short* wb_l1   = wb + 1 * WSZ;
    unsigned short* wb_r1   = wb + 2 * WSZ;
    unsigned short* wb_l2   = wb + 3 * WSZ;
    unsigned short* wb_r2   = wb + 4 * WSZ;
    unsigned short* wb_jk   = wb + 5 * WSZ;
    unsigned short* wb_post = wb + 6 * WSZ;      // 48 x 256 (zero padded)

    // 1) weights -> bf16
    k_cvt<<<WSZ / 256, 256, 0, stream>>>(W_pre, wb_pre, WSZ, WSZ);
    k_cvt<<<WSZ / 256, 256, 0, stream>>>(Wl1,   wb_l1,  WSZ, WSZ);
    k_cvt<<<WSZ / 256, 256, 0, stream>>>(Wr1,   wb_r1,  WSZ, WSZ);
    k_cvt<<<WSZ / 256, 256, 0, stream>>>(Wl2,   wb_l2,  WSZ, WSZ);
    k_cvt<<<WSZ / 256, 256, 0, stream>>>(Wr2,   wb_r2,  WSZ, WSZ);
    k_cvt<<<WSZ / 256, 256, 0, stream>>>(W_jk,  wb_jk,  WSZ, WSZ);
    k_cvt<<<(48 * DIM) / 256, 256, 0, stream>>>(W_post, wb_post, NCLS * DIM, 48 * DIM);

    const int nblk = NNODES / ROWS;              // 6250
    // 2) pre layer
    k_gemm_pre<<<nblk, 128, 0, stream>>>(x, wb_pre, b_pre, h);

    // 3) conv 1
    hipMemsetAsync(m, 0, actBytes, stream);
    k_gather<<<NEDGES / EPB, 256, 0, stream>>>(h, src, dst, m);
    k_gemm_conv<<<nblk, 128, 0, stream>>>(h, m, wb_l1, wb_r1, bl1);

    // 4) conv 2
    hipMemsetAsync(m, 0, actBytes, stream);
    k_gather<<<NEDGES / EPB, 256, 0, stream>>>(h, src, dst, m);
    k_gemm_conv<<<nblk, 128, 0, stream>>>(h, m, wb_l2, wb_r2, bl2);

    // 5) jk + post + log_softmax
    k_final<<<nblk, 128, 0, stream>>>(h, wb_jk, b_jk, wb_post, b_post, (float*)d_out);
}
